// CrossModalAttention_30013231464599
// MI455X (gfx1250) — matrix operand order repaired and checked by reading, not compiled, not run on hardware
//
#include <hip/hip_runtime.h>
#include <hip/hip_bf16.h>

// ---------------------------------------------------------------------------
// CrossModalAttention for MI455X (gfx1250, wave32)
//  - all GEMMs on v_wmma_f32_16x16x32_bf16 (f32 accumulate)
//  - activation tiles staged to LDS via Tensor Data Mover (TDM) with
//    hardware row padding (528B pitch -> conflict-free fragment reads),
//    cooperative-copy fallback if the TDM builtin is unavailable
// ---------------------------------------------------------------------------

#define BATCH 65536
#define HDIM  256
#define NHEAD 8
#define HDHD  32

typedef __attribute__((ext_vector_type(16))) __bf16 bf16x16;
typedef __attribute__((ext_vector_type(8)))  __bf16 bf16x8;
typedef __attribute__((ext_vector_type(4)))  __bf16 bf16x4;
typedef __attribute__((ext_vector_type(8)))  float  f32x8;
typedef __attribute__((ext_vector_type(4)))  float  f32x4;
typedef __attribute__((ext_vector_type(4)))  unsigned int u32x4;
typedef __attribute__((ext_vector_type(8)))  int    i32x8;
typedef __attribute__((ext_vector_type(4)))  int    i32x4;

#if defined(__HIP_DEVICE_COMPILE__) && defined(__gfx1250__) && \
    __has_builtin(__builtin_amdgcn_tensor_load_to_lds) && \
    __has_builtin(__builtin_amdgcn_s_wait_tensorcnt)
#define CM_TDM 1
#else
#define CM_TDM 0
#endif

// bf16 weight arena element offsets inside d_ws
constexpr size_t OWV  = 0;                    // Wv  [256,512]
constexpr size_t OWA  = OWV  + 512 * 256;     // Wa  [256,64]
constexpr size_t OWT  = OWA  + 64  * 256;     // Wt  [256,256]
constexpr size_t OWQ  = OWT  + 256 * 256;     // Wq
constexpr size_t OWK  = OWQ  + 256 * 256;     // Wk
constexpr size_t OWVL = OWK  + 256 * 256;     // Wval
constexpr size_t OWO  = OWVL + 256 * 256;     // Wo
constexpr size_t NWEL = OWO  + 256 * 256;
constexpr size_t WS_M_OFF   = (NWEL * 2 + 255) & ~(size_t)255;           // bytes
constexpr size_t WS_ATT_OFF = WS_M_OFF + (size_t)3 * BATCH * HDIM * 2;   // bytes

// LDS tile pitch: 264 bf16 = 528B per 256-elem row (16B pad every 512B,
// matches TDM pad_interval=128 DWORDs / pad_amount=4 DWORDs)
constexpr int LDSP = HDIM + 8;

// K3 dynamic LDS layout (bf16 elements / bytes)
constexpr int K3_MA_ELEMS   = 3 * 16 * LDSP;   // staged m tiles, 3 modalities
constexpr int K3_QKV_ELEMS  = 9 * 16 * LDSP;   // Q/K/V x 3 modalities
constexpr size_t K3_SC_OFF  = (size_t)(K3_MA_ELEMS + K3_QKV_ELEMS) * 2;  // bytes
constexpr size_t K3_LDS_BYTES = K3_SC_OFF + 2 * 6 * 16 * NHEAD * 4;

// K4 dynamic LDS layout (bytes)
constexpr size_t K4_ATT_BYTES  = (size_t)16 * LDSP * 2;                  // 8448
constexpr size_t K4_TILE_OFF   = K4_ATT_BYTES;                           // f32 [16][257]
constexpr size_t K4_PSUM_OFF   = K4_TILE_OFF + 16 * 257 * 4;
constexpr size_t K4_PSQ_OFF    = K4_PSUM_OFF + 256 * 4;
constexpr size_t K4_MU_OFF     = K4_PSQ_OFF + 256 * 4;
constexpr size_t K4_RS_OFF     = K4_MU_OFF + 16 * 4;
constexpr size_t K4_LDS_BYTES  = K4_RS_OFF + 16 * 4;

__constant__ int c_PQ[6] = {0, 0, 1, 1, 2, 2};
__constant__ int c_PK[6] = {1, 2, 0, 2, 0, 1};

// ---------------------------------------------------------------------------
// WMMA fragment helpers (layouts per CDNA5 ISA 7.12.2, wave32)
// ---------------------------------------------------------------------------
__device__ __forceinline__ bf16x16 frag_a_bf16(const __bf16* A, int lda, int lane) {
    const int sub = lane & 15, g = lane >> 4;
    const __bf16* p = A + (size_t)sub * lda + (g << 3);
    bf16x8 lo = *(const bf16x8*)p;
    bf16x8 hi = *(const bf16x8*)(p + 16);
    bf16x16 r;
#pragma unroll
    for (int i = 0; i < 8; ++i) { r[i] = lo[i]; r[i + 8] = hi[i]; }
    return r;
}

__device__ __forceinline__ bf16x16 frag_b(const __bf16* Wm, int ldw, int lane) {
    const int sub = lane & 15, g = lane >> 4;
    const __bf16* p = Wm + (size_t)sub * ldw + (g << 3);
    bf16x8 lo = *(const bf16x8*)p;
    bf16x8 hi = *(const bf16x8*)(p + 16);
    bf16x16 r;
#pragma unroll
    for (int i = 0; i < 8; ++i) { r[i] = lo[i]; r[i + 8] = hi[i]; }
    return r;
}

__device__ __forceinline__ f32x8 wmma_bf16(bf16x16 a, bf16x16 b, f32x8 c) {
    return __builtin_amdgcn_wmma_f32_16x16x32_bf16(false, a, false, b,
                                                   (short)0, c, false, false);
}

// ---------------------------------------------------------------------------
// TDM: async-DMA one contiguous 16x256 bf16 tile (4096 elems / 8KB) from
// global to LDS, inserting 16B of padding every 512B row -> 528B LDS pitch.
// One instruction per wave; completion tracked by TENSORcnt.
// (6-arg builtin form: g0, g1, g2, g3, g_extra, cpol)
// ---------------------------------------------------------------------------
#if CM_TDM
__device__ __forceinline__ void tdm_load_tile(const void* gsrc, unsigned ldsByteOff) {
    const unsigned long long ga = (unsigned long long)gsrc;
    u32x4 g0;
    g0[0] = 1u;                                              // count=1 (valid user D#)
    g0[1] = ldsByteOff;                                      // lds_addr
    g0[2] = (unsigned)ga;                                    // global_addr[31:0]
    g0[3] = (unsigned)((ga >> 32) & 0x1FFFFFFu) | (2u << 30);// global_addr[56:32] | type=2
    i32x8 g1;
    g1[0] = (int)((1u << 16)      // data_size = 2 bytes
                | (1u << 20)      // pad_enable
                | (6u << 22)      // pad_interval: 128 DWORDs (512B)
                | (3u << 25));    // pad_amount:   4 DWORDs  (16B)
    g1[1] = (int)(4096u << 16);   // tensor_dim0 = 4096 (bits[79:48] low half)
    g1[2] = (int)(1u << 16);      // tensor_dim0 hi=0 | tensor_dim1 = 1
    g1[3] = (int)(4096u << 16);   // tensor_dim1 hi=0 | tile_dim0 = 4096
    g1[4] = 0;                    // tile_dim1 = 0 (unused), tile_dim2 = 0
    g1[5] = 4096;                 // tensor_dim0_stride lo
    g1[6] = 0;
    g1[7] = 0;
    const i32x4 z4 = {0, 0, 0, 0};
    const i32x8 z8 = {0, 0, 0, 0, 0, 0, 0, 0};
    __builtin_amdgcn_tensor_load_to_lds(g0, g1, z4, z4, z8, 0);
}
#endif

// Fallback / generic cooperative copy of a contiguous 4096-elem bf16 tile
// into padded LDS (row pitch LDSP).
__device__ __forceinline__ void coop_copy_tile(const __bf16* src, __bf16* dst, int tid) {
    for (int i = tid * 4; i < 16 * HDIM; i += 256 * 4) {
        bf16x4 v = *(const bf16x4*)(src + i);
        *(bf16x4*)(dst + (i >> 8) * LDSP + (i & 255)) = v;
    }
}

// ---------------------------------------------------------------------------
// K1: f32 -> bf16 weight conversion
// ---------------------------------------------------------------------------
__global__ void cm_cvt_bf16(const float* __restrict__ src, __bf16* __restrict__ dst, int n) {
    int i = blockIdx.x * 256 + threadIdx.x;
    if (i < n) dst[i] = (__bf16)src[i];
}

// ---------------------------------------------------------------------------
// K2: modality projection  m[mod] = x[mod] @ W[mod]^T + b[mod]   (bf16 out)
// block: 256 thr (8 waves), 16 batch rows; input tile staged f32->bf16 in LDS
// ---------------------------------------------------------------------------
__global__ void cm_modproj(const float* __restrict__ xv, const float* __restrict__ xa,
                           const float* __restrict__ xt, const __bf16* __restrict__ wAll,
                           const float* __restrict__ bv, const float* __restrict__ ba,
                           const float* __restrict__ bt, __bf16* __restrict__ mOut) {
    extern __shared__ char smem[];
    __bf16* xs = (__bf16*)smem;                 // [16][K+8]

    const int mod  = blockIdx.y;
    const int row0 = blockIdx.x * 16;
    const int tid  = threadIdx.x;
    const int wave = tid >> 5;
    const int lane = tid & 31;
    const int sub  = lane & 15, g = lane >> 4;

    const float* x; int K; const __bf16* W; const float* bias;
    if (mod == 0)      { x = xv; K = 512; W = wAll + OWV; bias = bv; }
    else if (mod == 1) { x = xa; K = 64;  W = wAll + OWA; bias = ba; }
    else               { x = xt; K = 256; W = wAll + OWT; bias = bt; }
    const int KP = K + 8;

    // cooperative coalesced stage: 16 contiguous rows of f32 -> bf16 LDS tile
    const float* xsrc = x + (size_t)row0 * K;
    for (int i = tid * 4; i < 16 * K; i += 256 * 4) {
        f32x4 v = *(const f32x4*)(xsrc + i);
        bf16x4 b;
        b[0] = (__bf16)v[0]; b[1] = (__bf16)v[1];
        b[2] = (__bf16)v[2]; b[3] = (__bf16)v[3];
        *(bf16x4*)(xs + (i / K) * KP + (i % K)) = b;
    }
    __syncthreads();

#pragma unroll
    for (int t = 0; t < 2; ++t) {
        const int n0 = (wave * 2 + t) * 16;
        const __bf16* Wt_ = W + (size_t)n0 * K;
        f32x8 acc = {0.f, 0.f, 0.f, 0.f, 0.f, 0.f, 0.f, 0.f};
        for (int k0 = 0; k0 < K; k0 += 32) {
            if (k0 + 32 < K) __builtin_prefetch(Wt_ + k0 + 32, 0, 0);
            bf16x16 a = frag_a_bf16(xs + k0, KP, lane);
            bf16x16 b = frag_b(Wt_ + k0, K, lane);
            acc = wmma_bf16(a, b, acc);
        }
        const float bb = bias[n0 + sub];
        __bf16* o = mOut + (size_t)mod * BATCH * HDIM + (size_t)row0 * HDIM + n0;
#pragma unroll
        for (int v = 0; v < 8; ++v)
            o[(size_t)(v + 8 * g) * HDIM + sub] = (__bf16)(acc[v] + bb);
    }
}

// ---------------------------------------------------------------------------
// K3: TDM-stage m tiles -> 9 Q/K/V GEMMs into LDS -> per-head dot,
//     softmax over heads, weighting.  Writes attended (bf16) + weights (f32).
// ---------------------------------------------------------------------------
__global__ void cm_qkv_attn(const __bf16* __restrict__ wAll,
                            const float* __restrict__ bq, const float* __restrict__ bk,
                            const float* __restrict__ bval,
                            const __bf16* __restrict__ mIn,
                            __bf16* __restrict__ attOut, float* __restrict__ wOut) {
    extern __shared__ char smem[];
    __bf16* mA  = (__bf16*)smem;                 // [3][16][LDSP]
    __bf16* qkv = mA + K3_MA_ELEMS;              // [3 mat][3 mod][16][LDSP]
    float*  sc  = (float*)(smem + K3_SC_OFF);    // [6][16][8] scores
    float*  wts = sc + 6 * 16 * NHEAD;           // [6][16][8] weights

    const int row0 = blockIdx.x * 16;
    const int tid  = threadIdx.x;
    const int wave = tid >> 5;
    const int lane = tid & 31;
    const int sub  = lane & 15, g = lane >> 4;

    // ---- stage the three 16x256 m tiles into padded LDS ----
#if CM_TDM
    if (wave < 3) {
        const __bf16* src = mIn + (size_t)wave * BATCH * HDIM + (size_t)row0 * HDIM;
        tdm_load_tile(src, __builtin_amdgcn_groupstaticsize() +
                              (unsigned)(wave * 16 * LDSP * 2));
    }
    __builtin_amdgcn_s_wait_tensorcnt(0);
#else
    for (int mod = 0; mod < 3; ++mod)
        coop_copy_tile(mIn + (size_t)mod * BATCH * HDIM + (size_t)row0 * HDIM,
                       mA + mod * 16 * LDSP, tid);
#endif
    __syncthreads();

    const __bf16* Wmat[3] = {wAll + OWQ, wAll + OWK, wAll + OWVL};
    const float*  bmat[3] = {bq, bk, bval};

    // 9 GEMMs x 16 N-tiles = 144 tile jobs, wave-uniform distribution
    for (int job = wave; job < 144; job += 8) {
        const int mat = job / 48;
        const int mod = (job % 48) / 16;
        const int n0  = (job % 16) * 16;
        const __bf16* A = mA + mod * 16 * LDSP;
        const __bf16* W = Wmat[mat] + (size_t)n0 * HDIM;
        f32x8 acc = {0.f, 0.f, 0.f, 0.f, 0.f, 0.f, 0.f, 0.f};
#pragma unroll
        for (int k0 = 0; k0 < 256; k0 += 32) {
            if (k0 + 32 < 256) __builtin_prefetch(W + k0 + 32, 0, 0);
            bf16x16 a = frag_a_bf16(A + k0, LDSP, lane);
            bf16x16 b = frag_b(W + k0, HDIM, lane);
            acc = wmma_bf16(a, b, acc);
        }
        const float bb = bmat[mat][n0 + sub];
        __bf16* o = qkv + ((size_t)(mat * 3 + mod) * 16) * LDSP + n0;
#pragma unroll
        for (int v = 0; v < 8; ++v)
            o[(size_t)(v + 8 * g) * LDSP + sub] = (__bf16)(acc[v] + bb);
    }
    __syncthreads();

    // per-(pair,row,head) scalar dot of q,k over HD=32, scaled by 1/sqrt(32)
    for (int u = tid; u < 6 * 16 * NHEAD; u += 256) {
        const int p = u >> 7, r = (u >> 3) & 15, h = u & 7;
        const __bf16* q = qkv + ((0 * 3 + c_PQ[p]) * 16 + r) * LDSP + h * HDHD;
        const __bf16* k = qkv + ((1 * 3 + c_PK[p]) * 16 + r) * LDSP + h * HDHD;
        float s = 0.f;
#pragma unroll
        for (int d = 0; d < HDHD; ++d) s += (float)q[d] * (float)k[d];
        sc[u] = s * 0.17677669529663687f;   // 1/sqrt(32)
    }
    __syncthreads();

    // softmax across the 8 heads per (pair,row); emit attention-weights output
    if (tid < 96) {
        const int p = tid / 16, r = tid % 16;
        const float* s = sc + (p * 16 + r) * NHEAD;
        float mx = s[0];
#pragma unroll
        for (int h = 1; h < NHEAD; ++h) mx = fmaxf(mx, s[h]);
        float e[NHEAD], sum = 0.f;
#pragma unroll
        for (int h = 0; h < NHEAD; ++h) { e[h] = expf(s[h] - mx); sum += e[h]; }
        const float inv = 1.f / sum;
#pragma unroll
        for (int h = 0; h < NHEAD; ++h) {
            const float w = e[h] * inv;
            wts[(p * 16 + r) * NHEAD + h] = w;
            wOut[((size_t)p * BATCH + row0 + r) * NHEAD + h] = w;
        }
    }
    __syncthreads();

    // attended[p,r,n] = w[p,r,n/32] * V[pairK[p]][r][n]  -> bf16 global
    for (int i = tid; i < 6 * 16 * HDIM; i += 256) {
        const int p = i >> 12, r = (i >> 8) & 15, n = i & 255;
        const float w = wts[(p * 16 + r) * NHEAD + (n >> 5)];
        const float v = (float)qkv[((2 * 3 + c_PK[p]) * 16 + r) * LDSP + n];
        attOut[((size_t)p * BATCH + row0 + r) * HDIM + n] = (__bf16)(w * v);
    }
}

// ---------------------------------------------------------------------------
// K4: out = LN( attended @ Wo^T + bo + m[pairQ] ) * gamma + beta   (f32 out)
// attended tile TDM-staged into LDS; LN via two-stage LDS reduction
// ---------------------------------------------------------------------------
__global__ void cm_outproj_ln(const __bf16* __restrict__ wAll, const float* __restrict__ bo,
                              const __bf16* __restrict__ attIn, const __bf16* __restrict__ mIn,
                              const float* __restrict__ gamma, const float* __restrict__ beta,
                              float* __restrict__ out) {
    extern __shared__ char smem[];
    __bf16* attT = (__bf16*)smem;                           // [16][LDSP]
    float (*tile)[HDIM + 1] = (float (*)[HDIM + 1])(smem + K4_TILE_OFF);
    float* psum = (float*)(smem + K4_PSUM_OFF);             // [16][16]
    float* psq  = (float*)(smem + K4_PSQ_OFF);              // [16][16]
    float* smu  = (float*)(smem + K4_MU_OFF);               // [16]
    float* srs  = (float*)(smem + K4_RS_OFF);               // [16]

    const int p    = blockIdx.y;
    const int row0 = blockIdx.x * 16;
    const int tid  = threadIdx.x;
    const int wave = tid >> 5;
    const int lane = tid & 31;
    const int sub  = lane & 15, g = lane >> 4;

    const __bf16* Asrc = attIn + ((size_t)p * BATCH + row0) * HDIM;
#if CM_TDM
    if (wave == 0) tdm_load_tile(Asrc, __builtin_amdgcn_groupstaticsize());
    __builtin_amdgcn_s_wait_tensorcnt(0);
#else
    coop_copy_tile(Asrc, attT, tid);
#endif
    __syncthreads();

    const __bf16* W = wAll + OWO;
    const __bf16* R = mIn + ((size_t)c_PQ[p] * BATCH + row0) * HDIM;

#pragma unroll
    for (int t = 0; t < 2; ++t) {
        const int n0 = (wave * 2 + t) * 16;
        const __bf16* Wt_ = W + (size_t)n0 * HDIM;
        f32x8 acc = {0.f, 0.f, 0.f, 0.f, 0.f, 0.f, 0.f, 0.f};
#pragma unroll
        for (int k0 = 0; k0 < 256; k0 += 32) {
            if (k0 + 32 < 256) __builtin_prefetch(Wt_ + k0 + 32, 0, 0);
            bf16x16 a = frag_a_bf16(attT + k0, LDSP, lane);
            bf16x16 b = frag_b(Wt_ + k0, HDIM, lane);
            acc = wmma_bf16(a, b, acc);
        }
        const float bb = bo[n0 + sub];
#pragma unroll
        for (int v = 0; v < 8; ++v) {
            const int r = v + 8 * g;
            tile[r][n0 + sub] = acc[v] + bb + (float)R[(size_t)r * HDIM + n0 + sub];
        }
    }
    __syncthreads();

    // two-stage row reduction for LayerNorm stats
    {
        const int r = tid >> 4, s0 = (tid & 15) * 16;
        float s = 0.f, q = 0.f;
#pragma unroll
        for (int j = 0; j < 16; ++j) { const float x = tile[r][s0 + j]; s += x; q += x * x; }
        psum[r * 16 + (tid & 15)] = s;
        psq[r * 16 + (tid & 15)]  = q;
    }
    __syncthreads();
    if (tid < 16) {
        float s = 0.f, q = 0.f;
#pragma unroll
        for (int j = 0; j < 16; ++j) { s += psum[tid * 16 + j]; q += psq[tid * 16 + j]; }
        const float mu  = s * (1.f / 256.f);
        const float var = q * (1.f / 256.f) - mu * mu;
        smu[tid] = mu;
        srs[tid] = rsqrtf(var + 1e-5f);
    }
    __syncthreads();
    {
        const int r = tid >> 4, s0 = (tid & 15) * 16;
        const float mu = smu[r], ir = srs[r];
        float* o = out + ((size_t)p * BATCH + row0 + r) * HDIM;
#pragma unroll
        for (int j = 0; j < 16; ++j) {
            const int n = s0 + j;
            o[n] = (tile[r][n] - mu) * ir * gamma[n] + beta[n];
        }
    }
}

// ---------------------------------------------------------------------------
extern "C" void kernel_launch(void* const* d_in, const int* in_sizes, int n_in,
                              void* d_out, int out_size, void* d_ws, size_t ws_size,
                              hipStream_t stream) {
    const float* xv   = (const float*)d_in[0];
    const float* xa   = (const float*)d_in[1];
    const float* xt   = (const float*)d_in[2];
    const float* Wv   = (const float*)d_in[3];
    const float* bv   = (const float*)d_in[4];
    const float* Wa   = (const float*)d_in[5];
    const float* ba   = (const float*)d_in[6];
    const float* Wt   = (const float*)d_in[7];
    const float* bt   = (const float*)d_in[8];
    const float* Wq   = (const float*)d_in[9];
    const float* bq   = (const float*)d_in[10];
    const float* Wk   = (const float*)d_in[11];
    const float* bk   = (const float*)d_in[12];
    const float* Wval = (const float*)d_in[13];
    const float* bval = (const float*)d_in[14];
    const float* Wo   = (const float*)d_in[15];
    const float* bo   = (const float*)d_in[16];
    const float* gam  = (const float*)d_in[17];
    const float* bet  = (const float*)d_in[18];

    char* ws = (char*)d_ws;
    __bf16* wAll  = (__bf16*)ws;
    __bf16* mBF   = (__bf16*)(ws + WS_M_OFF);
    __bf16* attBF = (__bf16*)(ws + WS_ATT_OFF);

    float* outPtr = (float*)d_out;
    float* wtsPtr = outPtr + (size_t)6 * BATCH * HDIM;

    // K1: weights -> bf16
    auto cvt = [&](const float* s, __bf16* d, int n) {
        cm_cvt_bf16<<<(n + 255) / 256, 256, 0, stream>>>(s, d, n);
    };
    cvt(Wv,   wAll + OWV,  512 * 256);
    cvt(Wa,   wAll + OWA,  64 * 256);
    cvt(Wt,   wAll + OWT,  256 * 256);
    cvt(Wq,   wAll + OWQ,  256 * 256);
    cvt(Wk,   wAll + OWK,  256 * 256);
    cvt(Wval, wAll + OWVL, 256 * 256);
    cvt(Wo,   wAll + OWO,  256 * 256);

    // K2: modality projections (max LDS tile: 16 x (512+8) bf16)
    const size_t lds2 = (size_t)16 * (512 + 8) * 2;
    cm_modproj<<<dim3(BATCH / 16, 3), 256, lds2, stream>>>(xv, xa, xt, wAll, bv, ba, bt, mBF);

    // K3: fused QKV + attention
    cm_qkv_attn<<<dim3(BATCH / 16), 256, K3_LDS_BYTES, stream>>>(wAll, bq, bk, bval,
                                                                 mBF, attBF, wtsPtr);

    // K4: output projection + residual + LayerNorm
    cm_outproj_ln<<<dim3(BATCH / 16, 6), 256, K4_LDS_BYTES, stream>>>(wAll, bo, attBF, mBF,
                                                                      gam, bet, outPtr);
}
